// PGN_10213432230264
// MI455X (gfx1250) — compile-verified, matrix operand order checked by reading, not run on hardware
//
#include <hip/hip_runtime.h>

// ---------------------------------------------------------------------------
// PGN forward for MI455X (gfx1250, wave32, WMMA 16x16x32 f16 -> f32)
// ---------------------------------------------------------------------------
typedef __attribute__((ext_vector_type(16))) _Float16 v16h;
typedef __attribute__((ext_vector_type(8)))  float    v8f;

constexpr int kV = 50000, kE = 256, kH = 512, kB = 16, kS = 400, kT = 100;
constexpr int kG = 3 * kH;   // 1536
constexpr int kD = 2 * kH;   // 1024

__device__ __forceinline__ float sigf(float x) { return 1.f / (1.f + __expf(-x)); }

// A-fragment / B-fragment loader for v_wmma_f32_16x16x32_f16.
// Rows are K-contiguous with leading dimension ld (works for A[M][K] and W[N][K],
// since D = A @ W^T is symmetric in the two operands' storage).
// ISA 16-bit A layout: lanes 0-15 -> M=lane, K {0..7,16..23}; lanes 16-31 -> K {8..15,24..31}.
__device__ __forceinline__ v16h load_frag(const _Float16* base, int ld, int lane) {
  const int row  = lane & 15;
  const int koff = (lane & 16) ? 8 : 0;
  const _Float16* p = base + (size_t)row * ld + koff;
  v16h f;
#pragma unroll
  for (int i = 0; i < 8; ++i) f[i] = p[i];
#pragma unroll
  for (int i = 0; i < 8; ++i) f[i + 8] = p[i + 16];
  return f;
}

// ---------------------------------------------------------------------------
// Generic   C[M][N] = A[M][K] @ W[N][K]^T + bias[N]    (f16 in, f32 out)
// one wave per 16x16 tile, 8 waves / block
// ---------------------------------------------------------------------------
__global__ __launch_bounds__(256)
void wmma_gemm_atb(const _Float16* __restrict__ A, const _Float16* __restrict__ W,
                   const float* __restrict__ bias, float* __restrict__ C,
                   int M, int N, int K) {
  const int wave = (blockIdx.x * blockDim.x + threadIdx.x) >> 5;
  const int lane = threadIdx.x & 31;
  const int ntiles = N >> 4;
  const int total  = (M >> 4) * ntiles;
  if (wave >= total) return;
  const int mt = wave / ntiles;
  const int nt = wave - mt * ntiles;
  const _Float16* Ab = A + (size_t)(mt * 16) * K;
  const _Float16* Wb = W + (size_t)(nt * 16) * K;
  v8f acc = {};
  for (int k = 0; k < K; k += 32) {
    v16h a = load_frag(Ab + k, K, lane);
    v16h b = load_frag(Wb + k, K, lane);
    acc = __builtin_amdgcn_wmma_f32_16x16x32_f16(false, a, false, b, (short)0, acc,
                                                 false, false);
  }
  // C/D layout: VGPR i -> M = i (lanes 0-15) or 8+i (lanes 16-31); N = lane&15
  const int n  = (nt << 4) + (lane & 15);
  const int mb = (mt << 4) + ((lane & 16) ? 8 : 0);
  const float bv = bias ? bias[n] : 0.f;
#pragma unroll
  for (int i = 0; i < 8; ++i) C[(size_t)(mb + i) * N + n] = acc[i] + bv;
}

// ---------------------------------------------------------------------------
// small utility kernels
// ---------------------------------------------------------------------------
__global__ void cvt_f16_k(const float* __restrict__ in, _Float16* __restrict__ out, int n) {
  int i = blockIdx.x * blockDim.x + threadIdx.x;
  if (i < n) out[i] = (_Float16)in[i];
}

__global__ void zero_f32_k(float* p, int n) {
  int i = blockIdx.x * blockDim.x + threadIdx.x;
  if (i < n) p[i] = 0.f;
}

// xs[s][b][e] (f16) = enc_embed[clamp(src[b][s])][e]
__global__ void gather_enc_embed_k(const int* __restrict__ src, const float* __restrict__ emb,
                                   _Float16* __restrict__ xs) {
  int idx = blockIdx.x * blockDim.x + threadIdx.x;      // S*B*E = 1,638,400
  if (idx >= kS * kB * kE) return;
  int e  = idx & (kE - 1);
  int sb = idx >> 8;
  int b  = sb & (kB - 1);
  int s  = sb >> 4;
  int tok = src[b * kS + s];
  if (tok >= kV) tok = 1;                               // UNK
  xs[idx] = (_Float16)emb[(size_t)tok * kE + e];
}

// ---------------------------------------------------------------------------
// persistent GRU scan: one workgroup, 32 waves, h in LDS, gh via WMMA
// gi already contains x@Wih^T + bih.  gh gets bhh added at store.
// ---------------------------------------------------------------------------
__global__ __launch_bounds__(1024)
void gru_scan_k(const float* __restrict__ gi,      // [S][B][3H]  indexed by s_src
                const _Float16* __restrict__ Whh,  // [3H][H] f16
                const float* __restrict__ bhh,     // [3H]
                float* __restrict__ hs,            // [S][B][H]
                float* __restrict__ gh_ws,         // [B][3H] scratch (global)
                int rev) {
  __shared__ _Float16 h16[kB * kH];                 // 16 KB
  __shared__ float    hf [kB * kH];                 // 32 KB
  const int tid  = threadIdx.x;
  const int lane = tid & 31;
  const int wave = tid >> 5;
  for (int i = tid; i < kB * kH; i += 1024) { hf[i] = 0.f; h16[i] = (_Float16)0.f; }
  __syncthreads();

  for (int step = 0; step < kS; ++step) {
    const int s = rev ? (kS - 1 - step) : step;
    const float* gi_s = gi + (size_t)s * kB * kG;
    // gh = h @ Whh^T : 96 N-tiles, 3 per wave
#pragma unroll
    for (int t = 0; t < 3; ++t) {
      const int nt = wave * 3 + t;
      const _Float16* Wb = Whh + (size_t)(nt * 16) * kH;
      v8f acc = {};
      for (int k = 0; k < kH; k += 32) {
        v16h a = load_frag(h16 + k, kH, lane);     // from LDS
        v16h b = load_frag(Wb + k, kH, lane);      // from global (L2-resident)
        acc = __builtin_amdgcn_wmma_f32_16x16x32_f16(false, a, false, b, (short)0, acc,
                                                     false, false);
      }
      const int n  = (nt << 4) + (lane & 15);
      const int mb = (lane & 16) ? 8 : 0;
      const float bv = bhh[n];
#pragma unroll
      for (int i = 0; i < 8; ++i) gh_ws[(mb + i) * kG + n] = acc[i] + bv;
    }
    __threadfence_block();
    __syncthreads();
    // gates: 16*512 = 8192 elements, 8 per thread
#pragma unroll
    for (int k = 0; k < 8; ++k) {
      const int idx = tid + k * 1024;
      const int b = idx >> 9, j = idx & (kH - 1);
      const float* gib = gi_s  + b * kG;
      const float* ghb = gh_ws + b * kG;
      const float r  = sigf(gib[j] + ghb[j]);
      const float z  = sigf(gib[kH + j] + ghb[kH + j]);
      const float nn = tanhf(gib[2 * kH + j] + r * ghb[2 * kH + j]);
      const float hn = (1.f - z) * nn + z * hf[idx];
      hs[(size_t)s * kB * kH + idx] = hn;
      hf[idx]  = hn;
      h16[idx] = (_Float16)hn;
    }
    __syncthreads();
  }
}

// enc_out16[b][s][0:H]=hs_f, [H:2H]=hs_b   (hs stored [s][b][h])
__global__ void enc_concat_k(const float* __restrict__ hs_f, const float* __restrict__ hs_b,
                             _Float16* __restrict__ enc_out16) {
  int idx = blockIdx.x * blockDim.x + threadIdx.x;     // B*S*2H = 6,553,600
  if (idx >= kB * kS * kD) return;
  int d  = idx & (kD - 1);
  int bs = idx >> 10;
  int s  = bs % kS;
  int b  = bs / kS;
  float v = (d < kH) ? hs_f[((size_t)s * kB + b) * kH + d]
                     : hs_b[((size_t)s * kB + b) * kH + (d - kH)];
  enc_out16[idx] = (_Float16)v;
}

// enc_hidden[b][0:H]=hT_f (=hs_f at s=S-1), [H:2H]=hT_b (=hs_b stored at s=0)
__global__ void enc_hidden_k(const float* __restrict__ hs_f, const float* __restrict__ hs_b,
                             _Float16* __restrict__ eh16) {
  int idx = blockIdx.x * blockDim.x + threadIdx.x;     // 16*1024
  if (idx >= kB * kD) return;
  int d = idx & (kD - 1), b = idx >> 10;
  float v = (d < kH) ? hs_f[((size_t)(kS - 1) * kB + b) * kH + d]
                     : hs_b[(size_t)b * kH + (d - kH)];
  eh16[idx] = (_Float16)v;
}

// decoder-input embedding for step t
__global__ void dec_embed_k(const int* __restrict__ tgt, int t, const float* __restrict__ emb,
                            float* __restrict__ x, _Float16* __restrict__ x16) {
  int idx = blockIdx.x * blockDim.x + threadIdx.x;     // 16*256
  if (idx >= kB * kE) return;
  int e = idx & (kE - 1), b = idx >> 8;
  int tok = tgt[b * kT + t];
  if (tok >= kV) tok = 1;
  float v = emb[(size_t)tok * kE + e];
  x[idx] = v; x16[idx] = (_Float16)v;
}

// GRU cell gates for decoder (biases already in gi/gh)
__global__ void dec_gate_k(const float* __restrict__ gi, const float* __restrict__ gh,
                           const float* __restrict__ hprev,
                           float* __restrict__ hdec, _Float16* __restrict__ hdec16) {
  int idx = blockIdx.x * blockDim.x + threadIdx.x;     // 16*512
  if (idx >= kB * kH) return;
  int b = idx >> 9, j = idx & (kH - 1);
  const float* gib = gi + b * kG;
  const float* ghb = gh + b * kG;
  float r  = sigf(gib[j] + ghb[j]);
  float z  = sigf(gib[kH + j] + ghb[kH + j]);
  float nn = tanhf(gib[2 * kH + j] + r * ghb[2 * kH + j]);
  float hn = (1.f - z) * nn + z * hprev[idx];
  hdec[idx] = hn; hdec16[idx] = (_Float16)hn;
}

// e[b][s] = aV . tanh(Wh_enc[b][s] + s_part[b] + cov[b][s]*wc + abc) + abv
__global__ void attn_e_k(const float* __restrict__ Wh_enc, const float* __restrict__ s_part,
                         const float* __restrict__ cov, const float* __restrict__ aWc,
                         const float* __restrict__ abc, const float* __restrict__ aV,
                         const float* __restrict__ abv, float* __restrict__ e) {
  int idx = blockIdx.x * blockDim.x + threadIdx.x;     // B*S = 6400
  if (idx >= kB * kS) return;
  int b = idx / kS;
  const float* wh = Wh_enc + (size_t)idx * kH;
  const float* sp = s_part + b * kH;
  const float c = cov[idx];
  float acc = 0.f;
  for (int h = 0; h < kH; ++h)
    acc += aV[h] * tanhf(wh[h] + sp[h] + c * aWc[h] + abc[h]);
  e[idx] = acc + abv[0];
}

// masked softmax over S, coverage update, coverage loss (one wave per batch row)
__global__ __launch_bounds__(512)
void attn_softmax_k(const float* __restrict__ e, const float* __restrict__ mask,
                    float* __restrict__ attn, float* __restrict__ cov,
                    float* __restrict__ cov_loss) {
  const int b = threadIdx.x >> 5, lane = threadIdx.x & 31;
  if (b >= kB) return;
  const float* eb = e + b * kS;
  const float* mb = mask + b * kS;
  float mx = -1e30f;
  for (int s = lane; s < kS; s += 32) mx = fmaxf(mx, (mb[s] > 0.f) ? eb[s] : -1e30f);
  for (int o = 16; o > 0; o >>= 1) mx = fmaxf(mx, __shfl_xor(mx, o, 32));
  float sum = 0.f;
  for (int s = lane; s < kS; s += 32) {
    float v = (mb[s] > 0.f) ? __expf(eb[s] - mx) : 0.f;
    attn[b * kS + s] = v; sum += v;
  }
  for (int o = 16; o > 0; o >>= 1) sum += __shfl_xor(sum, o, 32);
  const float inv = 1.f / sum;
  float cl = 0.f;
  for (int s = lane; s < kS; s += 32) {
    float a  = attn[b * kS + s] * inv;
    float cn = cov[b * kS + s] + a;
    attn[b * kS + s] = a; cov[b * kS + s] = cn;
    cl += fminf(a, cn);                               // reference uses cov_new
  }
  for (int o = 16; o > 0; o >>= 1) cl += __shfl_xor(cl, o, 32);
  if (lane == 0) cov_loss[b] = cl;
}

// ctx[b][d] = sum_s attn[b][s] * enc_out[b][s][d]
__global__ void ctx_k(const float* __restrict__ attn, const _Float16* __restrict__ enc_out16,
                      float* __restrict__ ctx, _Float16* __restrict__ ctx16) {
  int idx = blockIdx.x * blockDim.x + threadIdx.x;     // 16*1024
  if (idx >= kB * kD) return;
  int b = idx >> 10, d = idx & (kD - 1);
  const _Float16* eb = enc_out16 + (size_t)b * kS * kD + d;
  const float* ab = attn + b * kS;
  float s = 0.f;
  for (int t = 0; t < kS; ++t) s += ab[t] * (float)eb[(size_t)t * kD];
  ctx[idx] = s; ctx16[idx] = (_Float16)s;
}

__global__ void concat_dh_k(const _Float16* __restrict__ a, const _Float16* __restrict__ h,
                            _Float16* __restrict__ out) {
  int idx = blockIdx.x * blockDim.x + threadIdx.x;     // 16*1024
  if (idx >= kB * kD) return;
  int b = idx >> 10, d = idx & (kD - 1);
  out[idx] = (d < kH) ? a[b * kH + d] : h[b * kH + (d - kH)];
}

// per-batch-row: softmax stats over V logits, p_gen, copy prob, NLL + coverage loss.
// tp = p_gen * p_vocab[tgt] + (1-p_gen) * sum_s attn[b][s]*(src[b][s]==tgt)
__global__ __launch_bounds__(256)
void loss_k(const float* __restrict__ logits, const int* __restrict__ tgt, int t,
            const float* __restrict__ attn, const int* __restrict__ src,
            const float* __restrict__ ctx_red, const float* __restrict__ hdec,
            const float* __restrict__ x, const float* __restrict__ gW,
            const float* __restrict__ gb, const float* __restrict__ cov_loss,
            float* __restrict__ loss_acc) {
  __shared__ float red[256];
  const int b = blockIdx.x, tid = threadIdx.x;
  const float* lg = logits + (size_t)b * kV;
  float mx = -1e30f;
  for (int i = tid; i < kV; i += 256) mx = fmaxf(mx, lg[i]);
  red[tid] = mx; __syncthreads();
  for (int s = 128; s > 0; s >>= 1) { if (tid < s) red[tid] = fmaxf(red[tid], red[tid + s]); __syncthreads(); }
  mx = red[0]; __syncthreads();
  float sum = 0.f;
  for (int i = tid; i < kV; i += 256) sum += __expf(lg[i] - mx);
  red[tid] = sum; __syncthreads();
  for (int s = 128; s > 0; s >>= 1) { if (tid < s) red[tid] += red[tid + s]; __syncthreads(); }
  const float Z = red[0]; __syncthreads();
  // p_gen score over concat [ctx_red(512), hdec(512), x(256)]
  float sc = 0.f;
  for (int i = tid; i < kD + kE; i += 256) {
    float v = (i < kH) ? ctx_red[b * kH + i]
             : (i < kD) ? hdec[b * kH + (i - kH)]
                        : x[b * kE + (i - kD)];
    sc += gW[i] * v;
  }
  red[tid] = sc; __syncthreads();
  for (int s = 128; s > 0; s >>= 1) { if (tid < s) red[tid] += red[tid + s]; __syncthreads(); }
  sc = red[0] + gb[0]; __syncthreads();
  const int tg = tgt[b * kT + t + 1];
  float cp = 0.f;
  for (int s = tid; s < kS; s += 256) if (src[b * kS + s] == tg) cp += attn[b * kS + s];
  red[tid] = cp; __syncthreads();
  for (int s = 128; s > 0; s >>= 1) { if (tid < s) red[tid] += red[tid + s]; __syncthreads(); }
  cp = red[0];
  if (tid == 0) {
    float pg = sigf(sc); pg = fminf(fmaxf(pg, 0.001f), 0.999f);
    float pv = (tg < kV) ? __expf(lg[tg] - mx) / Z : 0.f;
    float tp = pg * pv + (1.f - pg) * cp;
    float l  = -__logf(tp + 1e-9f) + cov_loss[b];     // LAMB = 1
    if (tg != 0) loss_acc[b] += l;
  }
}

__global__ void final_k(const float* __restrict__ loss_acc, const int* __restrict__ tgt,
                        float* __restrict__ out) {
  if (threadIdx.x == 0) {
    float m = 0.f;
    for (int b = 0; b < kB; ++b) {
      float len = -1.f;
      for (int t = 0; t < kT; ++t) len += (tgt[b * kT + t] != 0) ? 1.f : 0.f;
      m += loss_acc[b] / len;
    }
    out[0] = m / (float)kB;
  }
}

// ---------------------------------------------------------------------------
extern "C" void kernel_launch(void* const* d_in, const int* in_sizes, int n_in,
                              void* d_out, int out_size, void* d_ws, size_t ws_size,
                              hipStream_t stream) {
  (void)in_sizes; (void)n_in; (void)out_size; (void)ws_size;
  const int*   src_vecs  = (const int*)  d_in[0];
  const float* src_mask  = (const float*)d_in[1];
  const int*   tgt_vecs  = (const int*)  d_in[2];
  const float* enc_embed = (const float*)d_in[4];
  const float* dec_embed = (const float*)d_in[5];
  const float* eWih_f = (const float*)d_in[6];
  const float* eWhh_f = (const float*)d_in[7];
  const float* ebih_f = (const float*)d_in[8];
  const float* ebhh_f = (const float*)d_in[9];
  const float* eWih_b = (const float*)d_in[10];
  const float* eWhh_b = (const float*)d_in[11];
  const float* ebih_b = (const float*)d_in[12];
  const float* ebhh_b = (const float*)d_in[13];
  const float* dWih = (const float*)d_in[14];
  const float* dWhh = (const float*)d_in[15];
  const float* dbih = (const float*)d_in[16];
  const float* dbhh = (const float*)d_in[17];
  const float* aWh  = (const float*)d_in[18];
  const float* abh  = (const float*)d_in[19];
  const float* aWs  = (const float*)d_in[20];
  const float* absb = (const float*)d_in[21];
  const float* aWc  = (const float*)d_in[22];
  const float* abc  = (const float*)d_in[23];
  const float* aV   = (const float*)d_in[24];
  const float* abv  = (const float*)d_in[25];
  const float* rW   = (const float*)d_in[26];
  const float* rb   = (const float*)d_in[27];
  const float* gW   = (const float*)d_in[28];
  const float* gb   = (const float*)d_in[29];
  const float* fW   = (const float*)d_in[30];
  const float* fb   = (const float*)d_in[31];

  char* ws = (char*)d_ws;
  size_t off = 0;
  auto alloc = [&](size_t bytes) { size_t o = off; off += (bytes + 255) & ~(size_t)255; return o; };

  // f16 weight copies
  _Float16* fW16    = (_Float16*)(ws + alloc((size_t)kV * kD * 2));
  _Float16* eWihF16 = (_Float16*)(ws + alloc((size_t)kG * kE * 2));
  _Float16* eWhhF16 = (_Float16*)(ws + alloc((size_t)kG * kH * 2));
  _Float16* eWihB16 = (_Float16*)(ws + alloc((size_t)kG * kE * 2));
  _Float16* eWhhB16 = (_Float16*)(ws + alloc((size_t)kG * kH * 2));
  _Float16* dWih16  = (_Float16*)(ws + alloc((size_t)kG * kE * 2));
  _Float16* dWhh16  = (_Float16*)(ws + alloc((size_t)kG * kH * 2));
  _Float16* aWh16   = (_Float16*)(ws + alloc((size_t)kH * kD * 2));
  _Float16* aWs16   = (_Float16*)(ws + alloc((size_t)kH * kH * 2));
  _Float16* rW16    = (_Float16*)(ws + alloc((size_t)kH * kD * 2));
  // f16 activations
  _Float16* xs16     = (_Float16*)(ws + alloc((size_t)kS * kB * kE * 2));
  _Float16* encOut16 = (_Float16*)(ws + alloc((size_t)kB * kS * kD * 2));
  _Float16* encH16   = (_Float16*)(ws + alloc((size_t)kB * kD * 2));
  _Float16* hprev16  = (_Float16*)(ws + alloc((size_t)kB * kH * 2));
  _Float16* hdec16   = (_Float16*)(ws + alloc((size_t)kB * kH * 2));
  _Float16* ctx16    = (_Float16*)(ws + alloc((size_t)kB * kD * 2));
  _Float16* dh16     = (_Float16*)(ws + alloc((size_t)kB * kD * 2));
  _Float16* x16      = (_Float16*)(ws + alloc((size_t)kB * kE * 2));
  // f32 buffers
  float* giF     = (float*)(ws + alloc((size_t)kS * kB * kG * 4));
  float* giB     = (float*)(ws + alloc((size_t)kS * kB * kG * 4));
  float* hsF     = (float*)(ws + alloc((size_t)kS * kB * kH * 4));
  float* hsB     = (float*)(ws + alloc((size_t)kS * kB * kH * 4));
  float* ghF     = (float*)(ws + alloc((size_t)kB * kG * 4));
  float* ghB     = (float*)(ws + alloc((size_t)kB * kG * 4));
  float* WhEnc   = (float*)(ws + alloc((size_t)kB * kS * kH * 4));
  float* hprev   = (float*)(ws + alloc((size_t)kB * kH * 4));   // h carry (= ctx_red)
  float* giDec   = (float*)(ws + alloc((size_t)kB * kG * 4));
  float* ghDec   = (float*)(ws + alloc((size_t)kB * kG * 4));
  float* hdec    = (float*)(ws + alloc((size_t)kB * kH * 4));
  float* sPart   = (float*)(ws + alloc((size_t)kB * kH * 4));
  float* eBuf    = (float*)(ws + alloc((size_t)kB * kS * 4));
  float* attn    = (float*)(ws + alloc((size_t)kB * kS * 4));
  float* cov     = (float*)(ws + alloc((size_t)kB * kS * 4));
  float* covLoss = (float*)(ws + alloc((size_t)kB * 4));
  float* ctx     = (float*)(ws + alloc((size_t)kB * kD * 4));
  float* logits  = (float*)(ws + alloc((size_t)kB * kV * 4));
  float* lossAcc = (float*)(ws + alloc((size_t)kB * 4));
  float* xDec    = (float*)(ws + alloc((size_t)kB * kE * 4));

  auto cvt = [&](const float* in, _Float16* out, size_t n) {
    cvt_f16_k<<<(int)((n + 255) / 256), 256, 0, stream>>>(in, out, (int)n);
  };
  auto gemm = [&](const _Float16* A, const _Float16* W, const float* bias, float* C,
                  int M, int N, int K) {
    int waves = (M / 16) * (N / 16);
    wmma_gemm_atb<<<(waves + 7) / 8, 256, 0, stream>>>(A, W, bias, C, M, N, K);
  };

  // ---- one-time weight conversion (fW f16 copy is L2-resident: 102 MB < 192 MB) ----
  cvt(fW, fW16, (size_t)kV * kD);
  cvt(eWih_f, eWihF16, (size_t)kG * kE);  cvt(eWhh_f, eWhhF16, (size_t)kG * kH);
  cvt(eWih_b, eWihB16, (size_t)kG * kE);  cvt(eWhh_b, eWhhB16, (size_t)kG * kH);
  cvt(dWih, dWih16, (size_t)kG * kE);     cvt(dWhh, dWhh16, (size_t)kG * kH);
  cvt(aWh, aWh16, (size_t)kH * kD);       cvt(aWs, aWs16, (size_t)kH * kH);
  cvt(rW, rW16, (size_t)kH * kD);

  zero_f32_k<<<(kB * kS + 255) / 256, 256, 0, stream>>>(cov, kB * kS);
  zero_f32_k<<<1, 256, 0, stream>>>(lossAcc, kB);

  // ---- encoder ----
  gather_enc_embed_k<<<(kS * kB * kE + 255) / 256, 256, 0, stream>>>(src_vecs, enc_embed, xs16);
  gemm(xs16, eWihF16, ebih_f, giF, kS * kB, kG, kE);   // bulk input GEMM (fwd)
  gemm(xs16, eWihB16, ebih_b, giB, kS * kB, kG, kE);   // bulk input GEMM (bwd)
  gru_scan_k<<<1, 1024, 0, stream>>>(giF, eWhhF16, ebhh_f, hsF, ghF, 0);
  gru_scan_k<<<1, 1024, 0, stream>>>(giB, eWhhB16, ebhh_b, hsB, ghB, 1);
  enc_concat_k<<<(kB * kS * kD + 255) / 256, 256, 0, stream>>>(hsF, hsB, encOut16);
  enc_hidden_k<<<(kB * kD + 255) / 256, 256, 0, stream>>>(hsF, hsB, encH16);
  gemm(encH16, rW16, rb, hprev, kB, kH, kD);           // h_dec0
  cvt(hprev, hprev16, (size_t)kB * kH);
  gemm(encOut16, aWh16, abh, WhEnc, kB * kS, kH, kD);  // attention precompute

  // ---- decoder: T-1 sequential steps ----
  for (int t = 0; t < kT - 1; ++t) {
    dec_embed_k<<<(kB * kE + 255) / 256, 256, 0, stream>>>(tgt_vecs, t, dec_embed, xDec, x16);
    gemm(x16, dWih16, dbih, giDec, kB, kG, kE);
    gemm(hprev16, dWhh16, dbhh, ghDec, kB, kG, kH);
    dec_gate_k<<<(kB * kH + 255) / 256, 256, 0, stream>>>(giDec, ghDec, hprev, hdec, hdec16);
    gemm(hdec16, aWs16, absb, sPart, kB, kH, kH);
    attn_e_k<<<(kB * kS + 255) / 256, 256, 0, stream>>>(WhEnc, sPart, cov, aWc, abc, aV, abv, eBuf);
    attn_softmax_k<<<1, 512, 0, stream>>>(eBuf, src_mask, attn, cov, covLoss);
    ctx_k<<<(kB * kD + 255) / 256, 256, 0, stream>>>(attn, encOut16, ctx, ctx16);
    gemm(ctx16, rW16, rb, hprev, kB, kH, kD);          // ctx_red -> next h carry
    cvt(hprev, hprev16, (size_t)kB * kH);
    concat_dh_k<<<(kB * kD + 255) / 256, 256, 0, stream>>>(hprev16, hdec16, dh16);
    gemm(dh16, fW16, fb, logits, kB, kV, kD);          // the 16x50000x1024 projection
    loss_k<<<kB, 256, 0, stream>>>(logits, tgt_vecs, t, attn, src_vecs,
                                   hprev, hdec, xDec, gW, gb, covLoss, lossAcc);
  }
  final_k<<<1, 32, 0, stream>>>(lossAcc, tgt_vecs, (float*)d_out);
}
